// InstantNGP_67508295959092
// MI455X (gfx1250) — matrix operand order repaired and checked by reading, not compiled, not run on hardware
//
#include <hip/hip_runtime.h>
#include <hip/hip_bf16.h>
#include <stdint.h>

typedef __attribute__((ext_vector_type(16))) _Float16 v16h;
typedef __attribute__((ext_vector_type(8)))  float    v8f;

#define TLOG 19
#define TSIZE (1u << TLOG)
#define PTS_PER_BLOCK 128

__device__ __forceinline__ v8f wmma_f16(v16h a, v16h b, v8f c) {
  // D = A(16x32 f16) * B(32x16 f16) + C(16x16 f32)
  return __builtin_amdgcn_wmma_f32_16x16x32_f16(
      /*neg_a=*/false, a, /*neg_b=*/false, b,
      /*c_mod=*/(short)0, c, /*reuse_a=*/false, /*reuse_b=*/false);
}

// Load A fragment (16x32 f16, MxK) from row-major LDS buffer.
// ISA layout: lanes 0-15 M=lane; lanes 16-31 M=lane-16 with K offset +8.
// VGPR v holds K pair: (v&3)*2 + (v>>2)*16 + 8*halfsel.
__device__ __forceinline__ v16h load_a_frag(const _Float16* buf, int stride, int kbase) {
  const int l  = threadIdx.x & 31;
  const int m  = l & 15;
  const int hs = l >> 4;
  v16h a;
#pragma unroll
  for (int v = 0; v < 8; ++v) {
    const int k = kbase + ((v & 3) << 1) + ((v >> 2) << 4) + (hs << 3);
    a[2 * v]     = buf[m * stride + k];
    a[2 * v + 1] = buf[m * stride + k + 1];
  }
  return a;
}

// B fragments are pre-swizzled in LDS: [tile][lane][8 x u32] contiguous.
__device__ __forceinline__ v16h load_b_frag(const uint32_t* wfrag, int tile) {
  const int l = threadIdx.x & 31;
  return *(const v16h*)(wfrag + (size_t)(tile * 32 + l) * 8);
}

// D tile (16x16 f32): lane n=l&15, VGPR r -> row m = r + 8*(l>>4).
__device__ __forceinline__ void store_d_tile_relu(v8f d, _Float16* buf, int stride,
                                                  int col0, const float* bias) {
  const int l  = threadIdx.x & 31;
  const int n  = l & 15;
  const int hs = l >> 4;
  const float b = bias[col0 + n];
#pragma unroll
  for (int r = 0; r < 8; ++r) {
    const int m = r + (hs << 3);
    buf[m * stride + col0 + n] = (_Float16)fmaxf(d[r] + b, 0.0f);
  }
}

extern "C" __global__ void __launch_bounds__(256)
ngp_fused(const float* __restrict__ xin, const float* __restrict__ din,
          const float* __restrict__ table,
          const float* __restrict__ W1, const float* __restrict__ b1,
          const float* __restrict__ W2, const float* __restrict__ b2,
          const float* __restrict__ C1, const float* __restrict__ cb1,
          const float* __restrict__ C2, const float* __restrict__ cb2,
          const float* __restrict__ C3, const float* __restrict__ cb3,
          float* __restrict__ out_rgb, float* __restrict__ out_sigma, int npts) {
  // 20 B-tiles: W1:0-3  W2:4-5  C1:6-9  C2:10-17 (nb*2+kb)  C3:18-19
  __shared__ alignas(32) uint32_t wfrag[20 * 32 * 8];      // 20 KB
  __shared__ float sb1[64], sb2[16], scb1[64], scb2[64], scb3[16];
  __shared__ alignas(32) _Float16 actBuf[8][16 * 64];      // 16 KB (enc/h/c1/c2)
  __shared__ alignas(32) _Float16 catBuf[8][16 * 32];      //  8 KB (density|SH)

  const int tid = threadIdx.x;

  // ---- stage weights, pre-swizzled into WMMA B-fragment layout ----
  for (int idx = tid; idx < 20 * 32 * 8; idx += 256) {
    const int tile = idx >> 8;
    const int lane = (idx >> 3) & 31;
    const int v    = idx & 7;
    const float* W; int rowStride, nCols, kb, nb;
    if (tile < 4)       { W = W1; rowStride = 64; nCols = 64; kb = 0;         nb = tile; }
    else if (tile < 6)  { W = W2; rowStride = 16; nCols = 16; kb = tile - 4;  nb = 0; }
    else if (tile < 10) { W = C1; rowStride = 64; nCols = 64; kb = 0;         nb = tile - 6; }
    else if (tile < 18) { const int t = tile - 10; W = C2; rowStride = 64; nCols = 64; nb = t >> 1; kb = t & 1; }
    else                { W = C3; rowStride = 3;  nCols = 3;  kb = tile - 18; nb = 0; }
    const int hs = lane >> 4;
    const int n  = (lane & 15) + nb * 16;
    const int k  = kb * 32 + 2 * (v & 3) + 16 * (v >> 2) + 8 * hs;   // B layout: K pair 2(v&3) + 16(v>>2) + 8*halfsel
    const float f0 = (n < nCols) ? W[k * rowStride + n] : 0.0f;
    const float f1 = (n < nCols) ? W[(k + 1) * rowStride + n] : 0.0f;
    union { uint32_t u; _Float16 h[2]; } pk;
    pk.h[0] = (_Float16)f0; pk.h[1] = (_Float16)f1;
    wfrag[idx] = pk.u;
  }
  if (tid < 64) { sb1[tid] = b1[tid]; scb1[tid] = cb1[tid]; scb2[tid] = cb2[tid]; }
  if (tid < 16) { sb2[tid] = b2[tid]; scb3[tid] = (tid < 3) ? cb3[tid] : 0.0f; }
  __syncthreads();

  const int wave = tid >> 5;
  const int lane = tid & 31;
  const int m    = lane & 15;
  const int hs   = lane >> 4;
  const int p0   = blockIdx.x * PTS_PER_BLOCK + wave * 16;
  _Float16* act = actBuf[wave];
  _Float16* cat = catBuf[wave];
  const int p = min(p0 + m, npts - 1);

  // ---- hash-grid encode: lane owns point m at levels {2i + halfsel} ----
  {
    const float x0 = xin[3 * p + 0], x1 = xin[3 * p + 1], x2 = xin[3 * p + 2];
#pragma unroll
    for (int i = 0; i < 8; ++i) {
      const int lvl = 2 * i + hs;
      const float r = floorf(16.0f * expf(0.32346868426130781f * (float)lvl));
      const float px = x0 * r, py = x1 * r, pz = x2 * r;
      const float gx = floorf(px), gy = floorf(py), gz = floorf(pz);
      const float fx = px - gx, fy = py - gy, fz = pz - gz;
      const int ix = (int)gx, iy = (int)gy, iz = (int)gz;
      const float* tl = table + ((size_t)lvl << (TLOG + 1));
      float f0 = 0.0f, f1 = 0.0f;
#pragma unroll
      for (int c = 0; c < 8; ++c) {
        const int ox = (c >> 2) & 1, oy = (c >> 1) & 1, oz = c & 1;
        const uint32_t h = (uint32_t)(ix + ox)
                         ^ ((uint32_t)(iy + oy) * 2654435761u)
                         ^ ((uint32_t)(iz + oz) * 805459861u);
        const uint32_t idx = h & (TSIZE - 1u);
        const float2 f = *(const float2*)(tl + 2u * idx);   // L2-resident gather
        const float w = (ox ? fx : 1.0f - fx) * (oy ? fy : 1.0f - fy)
                      * (oz ? fz : 1.0f - fz);
        f0 += w * f.x; f1 += w * f.y;
      }
      act[m * 64 + 2 * lvl]     = (_Float16)f0;   // level-major, matches reshape
      act[m * 64 + 2 * lvl + 1] = (_Float16)f1;
    }
  }

  // ---- SH degree-4 into cat cols 16..31 ----
  {
    const float dx = din[3 * p], dy = din[3 * p + 1], dz = din[3 * p + 2];
    const float inv = rsqrtf(dx * dx + dy * dy + dz * dz);
    const float X = dx * inv, Y = dy * inv, Z = dz * inv;
    const float XX = X * X, YY = Y * Y, ZZ = Z * Z;
    float s[16];
    s[0]  = 0.28209479177387814f;
    s[1]  = -0.48860251190291987f * Y;
    s[2]  =  0.48860251190291987f * Z;
    s[3]  = -0.48860251190291987f * X;
    s[4]  =  1.0925484305920792f * X * Y;
    s[5]  = -1.0925484305920792f * Y * Z;
    s[6]  =  0.94617469575756f * ZZ - 0.31539156525252f;
    s[7]  = -1.0925484305920792f * X * Z;
    s[8]  =  0.5462742152960396f * (XX - YY);
    s[9]  =  0.5900435899266435f * Y * (-3.0f * XX + YY);
    s[10] =  2.890611442640554f * X * Y * Z;
    s[11] =  0.4570457994644657f * Y * (1.0f - 5.0f * ZZ);
    s[12] =  0.3731763325901154f * Z * (5.0f * ZZ - 3.0f);
    s[13] =  0.4570457994644657f * X * (1.0f - 5.0f * ZZ);
    s[14] =  1.445305721320277f * Z * (XX - YY);
    s[15] =  0.5900435899266435f * X * (-XX + 3.0f * YY);
    if (hs == 0) {
#pragma unroll
      for (int j = 0; j < 16; ++j) cat[m * 32 + 16 + j] = (_Float16)s[j];
    }
  }

  // ---- density layer 1: enc[16x32] @ W1[32x64] + b1, relu ----
  {
    const v16h a = load_a_frag(act, 64, 0);
    v8f h0 = {}, h1 = {}, h2 = {}, h3 = {};
    h0 = wmma_f16(a, load_b_frag(wfrag, 0), h0);
    h1 = wmma_f16(a, load_b_frag(wfrag, 1), h1);
    h2 = wmma_f16(a, load_b_frag(wfrag, 2), h2);
    h3 = wmma_f16(a, load_b_frag(wfrag, 3), h3);
    store_d_tile_relu(h0, act, 64, 0,  sb1);
    store_d_tile_relu(h1, act, 64, 16, sb1);
    store_d_tile_relu(h2, act, 64, 32, sb1);
    store_d_tile_relu(h3, act, 64, 48, sb1);
  }

  // ---- density layer 2: h[16x64] @ W2[64x16] + b2 ----
  v8f dens;
  {
    const v16h a0 = load_a_frag(act, 64, 0);
    const v16h a1 = load_a_frag(act, 64, 32);
    v8f t = {};
    t = wmma_f16(a0, load_b_frag(wfrag, 4), t);
    t = wmma_f16(a1, load_b_frag(wfrag, 5), t);
    dens = t;
  }
  {
    const float b = sb2[lane & 15];
#pragma unroll
    for (int r = 0; r < 8; ++r) dens[r] += b;
  }
  // sigma = exp(min(density[:,0], 15))  (column 0 lives in lanes 0 and 16)
  if ((lane & 15) == 0) {
#pragma unroll
    for (int r = 0; r < 8; ++r) {
      const int mm = r + (hs << 3);
      if (p0 + mm < npts) out_sigma[p0 + mm] = expf(fminf(dens[r], 15.0f));
    }
  }
  // density_out -> cat cols 0..15
  {
    const int n = lane & 15;
#pragma unroll
    for (int r = 0; r < 8; ++r) cat[(r + (hs << 3)) * 32 + n] = (_Float16)dens[r];
  }

  // ---- color layer 1: cat[16x32] @ C1[32x64] + cb1, relu ----
  {
    const v16h a = load_a_frag(cat, 32, 0);
    v8f t0 = {}, t1 = {}, t2 = {}, t3 = {};
    t0 = wmma_f16(a, load_b_frag(wfrag, 6), t0);
    t1 = wmma_f16(a, load_b_frag(wfrag, 7), t1);
    t2 = wmma_f16(a, load_b_frag(wfrag, 8), t2);
    t3 = wmma_f16(a, load_b_frag(wfrag, 9), t3);
    store_d_tile_relu(t0, act, 64, 0,  scb1);
    store_d_tile_relu(t1, act, 64, 16, scb1);
    store_d_tile_relu(t2, act, 64, 32, scb1);
    store_d_tile_relu(t3, act, 64, 48, scb1);
  }

  // ---- color layer 2: [16x64] @ C2[64x64] + cb2, relu ----
  {
    const v16h a0 = load_a_frag(act, 64, 0);
    const v16h a1 = load_a_frag(act, 64, 32);
    v8f t0 = {}, t1 = {}, t2 = {}, t3 = {};
    t0 = wmma_f16(a0, load_b_frag(wfrag, 10), t0);
    t0 = wmma_f16(a1, load_b_frag(wfrag, 11), t0);
    t1 = wmma_f16(a0, load_b_frag(wfrag, 12), t1);
    t1 = wmma_f16(a1, load_b_frag(wfrag, 13), t1);
    t2 = wmma_f16(a0, load_b_frag(wfrag, 14), t2);
    t2 = wmma_f16(a1, load_b_frag(wfrag, 15), t2);
    t3 = wmma_f16(a0, load_b_frag(wfrag, 16), t3);
    t3 = wmma_f16(a1, load_b_frag(wfrag, 17), t3);
    store_d_tile_relu(t0, act, 64, 0,  scb2);
    store_d_tile_relu(t1, act, 64, 16, scb2);
    store_d_tile_relu(t2, act, 64, 32, scb2);
    store_d_tile_relu(t3, act, 64, 48, scb2);
  }

  // ---- color layer 3: [16x64] @ C3[64x3] + cb3, sigmoid ----
  {
    const v16h a0 = load_a_frag(act, 64, 0);
    const v16h a1 = load_a_frag(act, 64, 32);
    v8f t = {};
    t = wmma_f16(a0, load_b_frag(wfrag, 18), t);
    t = wmma_f16(a1, load_b_frag(wfrag, 19), t);
    const int n = lane & 15;
    if (n < 3) {
      const float b = scb3[n];
#pragma unroll
      for (int r = 0; r < 8; ++r) {
        const int mm = r + (hs << 3);
        if (p0 + mm < npts) {
          const float v = t[r] + b;
          out_rgb[(size_t)(p0 + mm) * 3 + n] = 1.0f / (1.0f + expf(-v));
        }
      }
    }
  }
}

extern "C" void kernel_launch(void* const* d_in, const int* in_sizes, int n_in,
                              void* d_out, int out_size, void* d_ws, size_t ws_size,
                              hipStream_t stream) {
  const float* x   = (const float*)d_in[0];
  const float* d   = (const float*)d_in[1];
  const float* tab = (const float*)d_in[2];
  const float* W1  = (const float*)d_in[3];
  const float* b1  = (const float*)d_in[4];
  const float* W2  = (const float*)d_in[5];
  const float* b2  = (const float*)d_in[6];
  const float* C1  = (const float*)d_in[7];
  const float* cb1 = (const float*)d_in[8];
  const float* C2  = (const float*)d_in[9];
  const float* cb2 = (const float*)d_in[10];
  const float* C3  = (const float*)d_in[11];
  const float* cb3 = (const float*)d_in[12];
  const int npts = in_sizes[0] / 3;                 // x is [N,3]
  float* out_rgb   = (float*)d_out;                 // rgb first, then sigma
  float* out_sigma = out_rgb + (size_t)npts * 3;
  const int blocks = (npts + PTS_PER_BLOCK - 1) / PTS_PER_BLOCK;
  hipLaunchKernelGGL(ngp_fused, dim3(blocks), dim3(256), 0, stream,
                     x, d, tab, W1, b1, W2, b2, C1, cb1, C2, cb2, C3, cb3,
                     out_rgb, out_sigma, npts);
}
